// LSTM_77953656422681
// MI455X (gfx1250) — matrix operand order repaired and checked by reading, not compiled, not run on hardware
//
#include <hip/hip_runtime.h>

#define T_STEPS 512
#define BATCH   64
#define DDIM    512
#define HDIM    512
#define NWG     32
#define RS      520   // LDS row stride in halves: 1040B = 65*16B, conflict-free

typedef __attribute__((ext_vector_type(16))) _Float16     v16h;
typedef __attribute__((ext_vector_type(8)))  float        v8f;
typedef __attribute__((ext_vector_type(4)))  unsigned int u32x4;
typedef __attribute__((ext_vector_type(8)))  unsigned int u32x8;
typedef unsigned int uint32;

union Frag { u32x4 q[2]; v16h v; };   // 32B: one f16 WMMA A or B fragment/lane

// byte offset of h_s within the dynamic LDS block (after Wh_s + Wx_s)
#define LDS_H_OFF (128 * RS * 2 * 2)

struct LstmArgs {
  const float* Wx[4];   // W_xi, W_xf, W_xo, W_xc  [D, H] row-major
  const float* Wh[4];   // W_hi, W_hf, W_ho, W_hc  [H, H] row-major
  const float* bv[4];   // b_i, b_f, b_o, b_c      [H]
};

// ---------------------------------------------------------------------------
// Prep: X[b][t][d] (f32) -> Xh[t][b][d] (f16); zero h double-buffer + counters
// ---------------------------------------------------------------------------
__global__ __launch_bounds__(256) void lstm_prep(const float* __restrict__ X,
                                                 _Float16* __restrict__ Xh,
                                                 _Float16* __restrict__ hbuf,
                                                 int* __restrict__ counters) {
  const long stride = (long)gridDim.x * blockDim.x;
  const long N = (long)T_STEPS * BATCH * DDIM;
  for (long i = (long)blockIdx.x * blockDim.x + threadIdx.x; i < N; i += stride) {
    long d  = i & (DDIM - 1);
    long tb = i >> 9;              // / DDIM
    long b  = tb & (BATCH - 1);
    long t  = tb >> 6;             // / BATCH
    Xh[i] = (_Float16)X[(b * T_STEPS + t) * DDIM + d];
  }
  long i0 = (long)blockIdx.x * blockDim.x + threadIdx.x;
  if (i0 < 2L * BATCH * HDIM) hbuf[i0] = (_Float16)0.f;
  if (i0 < T_STEPS) counters[i0] = 0;
}

// ---------------------------------------------------------------------------
// Persistent LSTM scan. 32 WGs; WG w owns batch rows [mhalf*32,+32) and
// h-columns [colblk*32,+32) (all 4 gates). 8 waves x 2 tiles (shared A frag).
// h_{t-1} broadcast tile is pulled into LDS by the Tensor Data Mover.
// ---------------------------------------------------------------------------
__global__ __launch_bounds__(256) void lstm_persistent(
    const _Float16* __restrict__ Xh,   // [T][B][D] f16
    _Float16* __restrict__ hbuf,       // 2 x [B][H] f16 (double buffer)
    int* __restrict__ counters,        // [T]
    float* __restrict__ out,           // [B][T][H] f32
    LstmArgs args)
{
  extern __shared__ char smem[];
  _Float16* Wh_s = (_Float16*)smem;           // [128][RS]  W_h slice, [n][k]
  _Float16* Wx_s = Wh_s + 128 * RS;           // [128][RS]  W_x slice, [n][k]
  _Float16* h_s  = Wx_s + 128 * RS;           // [32][RS]   h_{t-1} rows (TDM dst)
  float*    g_s  = (float*)(h_s + 32 * RS);   // [16][16][16] gate tiles
  float*    c_s  = g_s + 16 * 256;            // [32][32] cell state (persistent)
  float*    b_s  = c_s + 1024;                // [128] bias slice

  const int tid    = threadIdx.x;
  const int w      = blockIdx.x;
  const int mhalf  = w & 1;                   // batch-row half
  const int colblk = w >> 1;                  // h-column block (32 cols)
  const int lane   = tid & 31;
  const int q      = tid >> 5;                // wave 0..7
  const int mt     = q >> 2;                  // M tile within half (0..1)
  const int gq     = q & 3;                   // gate (i,f,o,c)
  const int l      = lane & 15;
  const int hi     = lane >> 4;

  // ---- one-time init: weights + bias into LDS, zero cell state ----
  for (int idx = tid; idx < 128 * 512; idx += 256) {
    int n = idx >> 9, k = idx & 511;
    int g = n >> 5, j32 = n & 31;
    int gcol = colblk * 32 + j32;
    Wh_s[n * RS + k] = (_Float16)args.Wh[g][k * HDIM + gcol];
    Wx_s[n * RS + k] = (_Float16)args.Wx[g][k * HDIM + gcol];
  }
  if (tid < 128) {
    int g = tid >> 5, j32 = tid & 31;
    b_s[tid] = args.bv[g][colblk * 32 + j32];
  }
  for (int idx = tid; idx < 1024; idx += 256) c_s[idx] = 0.f;
  __syncthreads();

  // 16B-unit bases (row stride RS/8 = 65 u32x4 per row)
  const u32x4* WhP = (const u32x4*)Wh_s;
  const u32x4* WxP = (const u32x4*)Wx_s;
  const u32x4* hP  = (const u32x4*)h_s;
  const int abase  = (mt * 16 + l) * 65 + hi;            // A: +kb*4, chunks +0/+2
  const int bbase0 = (gq * 32 + l) * 65 + hi;            // B tile0: +kb*4, +0/+2
  const int bbase1 = (gq * 32 + 16 + l) * 65 + hi;       // B tile1

  // TDM D# group1 (constant): data_size=2B, pad_enable, pad every 256 DW by
  // 4 DW (-> 520-half LDS rows), tensor 512x64, tile 512x32, dim0 stride 512.
  u32x8 tdm_g1;
  tdm_g1[0] = (1u << 16) | (1u << 20) | (7u << 22) | (3u << 25);
  tdm_g1[1] = (512u << 16);          // tensor_dim0 = 512 (bits 79:48)
  tdm_g1[2] = (64u << 16);           // tensor_dim1 = 64  (bits 111:80)
  tdm_g1[3] = (512u << 16);          // tile_dim0 = 512   (bits 127:112)
  tdm_g1[4] = 32u;                   // tile_dim1 = 32    (bits 143:128)
  tdm_g1[5] = 512u;                  // tensor_dim0_stride = 512 (bits 207:160)
  tdm_g1[6] = 0u;
  tdm_g1[7] = 0u;

  for (int t = 0; t < T_STEPS; ++t) {
    // ---- wait until h_{t-1} is fully published (all 32 WGs) ----
    if (t > 0 && tid == 0) {
      while (__hip_atomic_load(&counters[t - 1], __ATOMIC_ACQUIRE,
                               __HIP_MEMORY_SCOPE_AGENT) < NWG) {
        __builtin_amdgcn_s_sleep(1);
      }
    }
    __syncthreads();

    // ---- TDM: DMA h_{t-1} rows [mhalf*32,+32) x 512 into padded LDS ----
    if (q == 0) {
      unsigned long long gaddr =
          (unsigned long long)(hbuf + ((t + 1) & 1) * (BATCH * HDIM)
                                    + mhalf * 32 * HDIM);
      u32x4 g0;
      g0[0] = 1u;                                   // count=1 (valid user D#)
      g0[1] = (unsigned)LDS_H_OFF;                  // lds_addr (bytes)
      g0[2] = (unsigned)(gaddr & 0xFFFFFFFFu);      // global_addr[31:0]
      g0[3] = (unsigned)((gaddr >> 32) & 0x01FFFFFFu) | (2u << 30);  // type=2
      asm volatile("tensor_load_to_lds %0, %1 scope:SCOPE_DEV"
                   :: "s"(g0), "s"(tdm_g1) : "memory");
#if __has_builtin(__builtin_amdgcn_s_wait_tensorcnt)
      __builtin_amdgcn_s_wait_tensorcnt(0);
#else
      asm volatile("s_wait_tensorcnt 0x0" ::: "memory");
#endif
    }
    if (t + 1 < T_STEPS) {   // prefetch next step's x tile toward L2
      const char* pf = (const char*)(Xh + ((long)(t + 1) * BATCH + mhalf * 32) * DDIM);
      __builtin_prefetch(pf + (size_t)tid * 128, 0, 1);
    }
    __syncthreads();

    v8f acc0 = {0.f, 0.f, 0.f, 0.f, 0.f, 0.f, 0.f, 0.f};
    v8f acc1 = {0.f, 0.f, 0.f, 0.f, 0.f, 0.f, 0.f, 0.f};

    // ---- phase 1: h_{t-1} @ W_h (A from LDS), software-pipelined ----
    {
      Frag a, b0, b1;
      a.q[0]  = hP[abase];       a.q[1]  = hP[abase + 2];
      b0.q[0] = WhP[bbase0];     b0.q[1] = WhP[bbase0 + 2];
      b1.q[0] = WhP[bbase1];     b1.q[1] = WhP[bbase1 + 2];
#pragma unroll
      for (int kb = 0; kb < 16; ++kb) {
        Frag an = a, bn0 = b0, bn1 = b1;
        if (kb < 15) {
          int o = (kb + 1) * 4;
          an.q[0]  = hP[abase + o];       an.q[1]  = hP[abase + o + 2];
          bn0.q[0] = WhP[bbase0 + o];     bn0.q[1] = WhP[bbase0 + o + 2];
          bn1.q[0] = WhP[bbase1 + o];     bn1.q[1] = WhP[bbase1 + o + 2];
        }
        acc0 = __builtin_amdgcn_wmma_f32_16x16x32_f16(false, a.v, false, b0.v,
                                                      (short)0, acc0, false, false);
        acc1 = __builtin_amdgcn_wmma_f32_16x16x32_f16(false, a.v, false, b1.v,
                                                      (short)0, acc1, false, false);
        a = an; b0 = bn0; b1 = bn1;
      }
    }
    // ---- phase 2: x_t @ W_x (A straight from global Xh, L2-hot) ----
    {
      const u32x4* xP = (const u32x4*)(Xh + ((long)t * BATCH) * DDIM);
      const int xbase = (mhalf * 32 + mt * 16 + l) * 64 + hi;  // 64 u32x4 per row
      Frag a, b0, b1;
      a.q[0]  = xP[xbase];       a.q[1]  = xP[xbase + 2];
      b0.q[0] = WxP[bbase0];     b0.q[1] = WxP[bbase0 + 2];
      b1.q[0] = WxP[bbase1];     b1.q[1] = WxP[bbase1 + 2];
#pragma unroll
      for (int kb = 0; kb < 16; ++kb) {
        Frag an = a, bn0 = b0, bn1 = b1;
        if (kb < 15) {
          int o = (kb + 1) * 4;
          an.q[0]  = xP[xbase + o];       an.q[1]  = xP[xbase + o + 2];
          bn0.q[0] = WxP[bbase0 + o];     bn0.q[1] = WxP[bbase0 + o + 2];
          bn1.q[0] = WxP[bbase1 + o];     bn1.q[1] = WxP[bbase1 + o + 2];
        }
        acc0 = __builtin_amdgcn_wmma_f32_16x16x32_f16(false, a.v, false, b0.v,
                                                      (short)0, acc0, false, false);
        acc1 = __builtin_amdgcn_wmma_f32_16x16x32_f16(false, a.v, false, b1.v,
                                                      (short)0, acc1, false, false);
        a = an; b0 = bn0; b1 = bn1;
      }
    }

    // D layout: VGPR r -> M = r + hi*8, N = l.  Tiles q*2 and q*2+1.
#pragma unroll
    for (int r = 0; r < 8; ++r) {
      g_s[(q * 2 + 0) * 256 + (r + hi * 8) * 16 + l] = acc0[r];
      g_s[(q * 2 + 1) * 256 + (r + hi * 8) * 16 + l] = acc1[r];
    }
    __syncthreads();

    // ---- gates + state update (32 rows x 32 cols per WG) ----
    _Float16* hw = hbuf + (t & 1) * (BATCH * HDIM);
    for (int e = tid; e < 1024; e += 256) {
      int br = e >> 5, j32 = e & 31;
      int mtt = br >> 4, rr = br & 15;
      int jblk = j32 >> 4, j = j32 & 15;
      float gi = g_s[((mtt * 4 + 0) * 2 + jblk) * 256 + rr * 16 + j] + b_s[j32];
      float gf = g_s[((mtt * 4 + 1) * 2 + jblk) * 256 + rr * 16 + j] + b_s[32 + j32];
      float go = g_s[((mtt * 4 + 2) * 2 + jblk) * 256 + rr * 16 + j] + b_s[64 + j32];
      float gc = g_s[((mtt * 4 + 3) * 2 + jblk) * 256 + rr * 16 + j] + b_s[96 + j32];
      float I  = 1.f / (1.f + expf(-gi));
      float F  = 1.f / (1.f + expf(-gf));
      float O  = 1.f / (1.f + expf(-go));
      float Cd = tanhf(gc);
      float c  = F * c_s[e] + I * Cd;
      c_s[e]   = c;
      float hv = O * tanhf(c);
      int b    = mhalf * 32 + br;
      int hcol = colblk * 32 + j32;
      out[((long)b * T_STEPS + t) * HDIM + hcol] = hv;
      hw[b * HDIM + hcol] = (_Float16)hv;
    }
    __threadfence();
    __syncthreads();
    if (tid == 0)
      __hip_atomic_fetch_add(&counters[t], 1, __ATOMIC_RELEASE,
                             __HIP_MEMORY_SCOPE_AGENT);
  }
}

// ---------------------------------------------------------------------------
extern "C" void kernel_launch(void* const* d_in, const int* in_sizes, int n_in,
                              void* d_out, int out_size, void* d_ws, size_t ws_size,
                              hipStream_t stream) {
  const float* X = (const float*)d_in[0];
  LstmArgs a;
  for (int g = 0; g < 4; ++g) {
    a.Wx[g] = (const float*)d_in[1 + 3 * g];
    a.Wh[g] = (const float*)d_in[2 + 3 * g];
    a.bv[g] = (const float*)d_in[3 + 3 * g];
  }

  // ws layout: Xh f16 [T][B][D] (32MB) | hbuf f16 2x[B][H] (128KB) | counters
  _Float16* Xh       = (_Float16*)d_ws;
  _Float16* hbuf     = Xh + (size_t)T_STEPS * BATCH * DDIM;
  int*      counters = (int*)(hbuf + 2 * BATCH * HDIM);
  float*    out      = (float*)d_out;

  lstm_prep<<<2048, 256, 0, stream>>>(X, Xh, hbuf, counters);

  // LDS: 2*128*RS*2 + 32*RS*2 + (16*256 + 1024 + 128)*4 = 320,512 B (< 320KB)
  const int smem_bytes = 128 * RS * 2 * 2 + 32 * RS * 2
                       + (16 * 256 + 1024 + 128) * 4;
  (void)hipFuncSetAttribute((const void*)lstm_persistent,
                            hipFuncAttributeMaxDynamicSharedMemorySize,
                            smem_bytes);
  lstm_persistent<<<NWG, 256, smem_bytes, stream>>>(Xh, hbuf, counters, out, a);
}
